// EdgeRolandGNN_44117904065163
// MI455X (gfx1250) — matrix operand order repaired and checked by reading, not compiled, not run on hardware
//
#include <hip/hip_runtime.h>
#include <hip/hip_bf16.h>

// ---------------------------------------------------------------------------
// EdgeRolandGNN forward on gfx1250 (MI455X).
// Dense matmuls: v_wmma_f32_16x16x32_f16 on pre-converted f16 operands,
// 128x64 block tile, 8 waves, 4 WMMA/wave per 32-wide K step.
// A tile staged by GLOBAL_LOAD_ASYNC_TO_LDS_B128 (ASYNCcnt, no VGPRs);
// B tile staged transposed via VGPR (ds_load_2addr_b64 fragments).
// GCN scatter uses float atomics; GRU gates / finalize are elementwise.
// ---------------------------------------------------------------------------

typedef __attribute__((ext_vector_type(16))) _Float16 v16h;
typedef __attribute__((ext_vector_type(8)))  float    v8f;

#define GNN_N   50000
#define GNN_E   800000
#define GNN_EQ  200000

constexpr int LDA = 40;   // halves; 80B row stride (16B-aligned async chunks)
constexpr int LDBT = 36;  // halves; 72B row stride (8B-aligned b64 fragment loads)

__device__ __forceinline__ float lrelu(float v) { return v > 0.0f ? v : 0.01f * v; }
__device__ __forceinline__ float sigmoidf(float v) { return 1.0f / (1.0f + __expf(-v)); }

__device__ __forceinline__ unsigned lds_off(const void* p) {
    return (unsigned)(uintptr_t)p;           // low 32 bits of the pointer give the LDS offset
}

__device__ __forceinline__ void async_cp16(unsigned lds, const _Float16* g) {
    asm volatile("global_load_async_to_lds_b128 %0, %1, off"
                 :: "v"(lds), "v"(g) : "memory");
}
__device__ __forceinline__ void wait_async0() {
    asm volatile("s_wait_asynccnt 0x0" ::: "memory");
}

union h8 { uint4 u; _Float16 h[8]; };

// ---------------------------------------------------------------------------
// WMMA GEMM: C[M,Nout] = act(A16[M,K] @ W16[K,Nout] + bias)
// block = 256 threads (8 waves), tile 128(M) x 64(N), K step 32,
// double-buffered LDS, one barrier per step.
// ---------------------------------------------------------------------------
template<int K>
__global__ __launch_bounds__(256)
void gemm_wmma_f16(const _Float16* __restrict__ A, const _Float16* __restrict__ W,
                   const float* __restrict__ bias, float* __restrict__ C,
                   int M, int Nout, int act)
{
    __shared__ _Float16 As[2][128 * LDA];    // 128 rows x 32 k (f16)
    __shared__ _Float16 Bt[2][64 * LDBT];    // transposed: 64 n-cols x 32 k (f16)

    const int tid  = threadIdx.x;
    const int wave = tid >> 5;
    const int lane = tid & 31;
    const int hi   = lane >> 4;              // half-wave select
    const int l16  = lane & 15;
    const int M0   = blockIdx.x * 128;
    const int N0   = blockIdx.y * 64;
    const int bkk  = tid >> 3;               // B chunk: k row 0..31
    const int bc8  = (tid & 7) << 3;         // B chunk: n col base (8 halves)

    // async stage of A panel p (128x32 halves) into buffer buf: 2 issues/thread
    auto stage_a_async = [&](int p, int buf) {
        const int k0 = p * 32;
        #pragma unroll
        for (int j = 0; j < 2; ++j) {
            int slot = j * 256 + tid;        // 512 16B chunks
            int r = slot >> 2;
            int c = (slot & 3) << 3;         // halves
            int gr = M0 + r; if (gr > M - 1) gr = M - 1;   // clamp; epilogue guards
            async_cp16(lds_off(&As[buf][r * LDA + c]),
                       A + (size_t)gr * K + k0 + c);
        }
    };
    // B panel: one uint4 (8 halves of row k0+bkk) per thread
    auto stage_b_issue = [&](int p) -> uint4 {
        return *(const uint4*)(W + (size_t)(p * 32 + bkk) * Nout + N0 + bc8);
    };
    auto stage_b_commit = [&](int buf, uint4 v) {
        h8 q; q.u = v;
        #pragma unroll
        for (int t = 0; t < 8; ++t)          // transposed scatter Bt[n][k]
            Bt[buf][(bc8 + t) * LDBT + bkk] = q.h[t];
    };

    v8f acc[4] = {};

    stage_a_async(0, 0);
    stage_b_commit(0, stage_b_issue(0));
    wait_async0();
    __syncthreads();

    constexpr int NSTEP = K / 32;
    #pragma unroll
    for (int s = 0; s < NSTEP; ++s) {
        const int cur = s & 1;

        uint4 vb;
        if (s + 1 < NSTEP) {                 // overlap next panel copy with WMMAs
            stage_a_async(s + 1, cur ^ 1);
            vb = stage_b_issue(s + 1);
        }
        if (s + 2 < NSTEP) {                 // prefetch panel s+2 of A
            int pr = M0 + (tid & 127); if (pr > M - 1) pr = M - 1;
            __builtin_prefetch(A + (size_t)pr * K + (s + 2) * 32, 0, 1);
        }

        // ---- A fragment (ISA 16-bit A layout, wave32): two aligned b128 ----
        v16h a;
        const _Float16* arow = &As[cur][(wave * 16 + l16) * LDA + hi * 8];
        #pragma unroll
        for (int i = 0; i < 8; ++i) { a[i] = arow[i]; a[i + 8] = arow[16 + i]; }

        #pragma unroll
        for (int nt = 0; nt < 4; ++nt) {     // B fragment: k = 8h+i and 16+8h+i (matches A)
            v16h b;
            const _Float16* brow = &Bt[cur][(nt * 16 + l16) * LDBT + hi * 8];
            #pragma unroll
            for (int i = 0; i < 8; ++i) { b[i] = brow[i]; b[i + 8] = brow[16 + i]; }
            acc[nt] = __builtin_amdgcn_wmma_f32_16x16x32_f16(false, a, false, b,
                                                             (short)0, acc[nt],
                                                             false, false);
        }

        if (s + 1 < NSTEP) {
            stage_b_commit(cur ^ 1, vb);     // LDS stores after compute
            wait_async0();                   // drain own async copies
        }
        __syncthreads();
    }

    // ---- epilogue: C/D layout VGPR r -> M = hi*8 + r, N = lane%16 ----
    #pragma unroll
    for (int nt = 0; nt < 4; ++nt) {
        int gn = N0 + nt * 16 + l16;
        float bb = bias ? bias[gn] : 0.0f;
        #pragma unroll
        for (int r = 0; r < 8; ++r) {
            int gm = M0 + wave * 16 + hi * 8 + r;
            if (gm < M) {
                float v = acc[nt][r] + bb;
                if (act == 1) v = lrelu(v);
                C[(size_t)gm * Nout + gn] = v;
            }
        }
    }
}

// ---------------------------------------------------------------------------
// f32 -> f16 conversion (one float4 per thread, packed cvt + b64 store)
// ---------------------------------------------------------------------------
__global__ void cvt_f16_kernel(const float* __restrict__ in,
                               _Float16* __restrict__ out, int n4) {
    int i = blockIdx.x * blockDim.x + threadIdx.x;
    if (i >= n4) return;
    float4 v = ((const float4*)in)[i];
    auto p0 = __builtin_amdgcn_cvt_pkrtz(v.x, v.y);
    auto p1 = __builtin_amdgcn_cvt_pkrtz(v.z, v.w);
    *(decltype(p0)*)(out + (size_t)i * 4)     = p0;
    *(decltype(p1)*)(out + (size_t)i * 4 + 2) = p1;
}

// ---------------------------------------------------------------------------
// GCN helpers
// ---------------------------------------------------------------------------
__global__ void fill_kernel(float* __restrict__ p, float v, int n) {
    int i = blockIdx.x * blockDim.x + threadIdx.x;
    if (i < n) p[i] = v;
}

__global__ void deg_count_kernel(const long long* __restrict__ ei,
                                 float* __restrict__ deg, int E) {
    int e = blockIdx.x * blockDim.x + threadIdx.x;
    if (e < E) {
        int col = (int)ei[E + e];
        atomicAdd(&deg[col], 1.0f);
    }
}

__global__ void rsqrt_kernel(float* __restrict__ deg, int n) {
    int i = blockIdx.x * blockDim.x + threadIdx.x;
    if (i < n) {
        float d = deg[i];
        deg[i] = d > 0.0f ? rsqrtf(d) : 0.0f;
    }
}

// scatter: agg[col,f] += hw[row,f] * dinv[row]*dinv[col]; 4 features/thread
__global__ void gcn_scatter_kernel(const long long* __restrict__ ei,
                                   const float* __restrict__ hw,
                                   const float* __restrict__ dinv,
                                   float* __restrict__ agg, int E, int F) {
    int gid = blockIdx.x * blockDim.x + threadIdx.x;
    int tpe = F >> 2;
    int e = gid / tpe;
    if (e >= E) return;
    int f = (gid % tpe) << 2;
    int row = (int)ei[e];
    int col = (int)ei[E + e];
    float norm = dinv[row] * dinv[col];
    const float* src = hw + (size_t)row * F + f;
    float* dst = agg + (size_t)col * F + f;
    atomicAdd(dst + 0, src[0] * norm);
    atomicAdd(dst + 1, src[1] * norm);
    atomicAdd(dst + 2, src[2] * norm);
    atomicAdd(dst + 3, src[3] * norm);
}

// c = lrelu(agg + hw*dinv^2 (self loop) + bias)
__global__ void gcn_finalize_kernel(const float* __restrict__ agg,
                                    const float* __restrict__ hw,
                                    const float* __restrict__ dinv,
                                    const float* __restrict__ bias,
                                    float* __restrict__ c, int N, int F) {
    int gid = blockIdx.x * blockDim.x + threadIdx.x;
    if (gid >= N * F) return;
    int n = gid / F;
    int f = gid - n * F;
    float d = dinv[n];
    c[gid] = lrelu(agg[gid] + hw[gid] * d * d + bias[f]);
}

// ---------------------------------------------------------------------------
// GRU cell gates (gate order r, z, n)
// ---------------------------------------------------------------------------
__global__ void gru_kernel(const float* __restrict__ gi, const float* __restrict__ gh,
                           const float* __restrict__ prev, float* __restrict__ out,
                           int N, int H) {
    int gid = blockIdx.x * blockDim.x + threadIdx.x;
    if (gid >= N * H) return;
    int n = gid / H;
    int j = gid - n * H;
    size_t b = (size_t)n * 3 * H;
    float ir = gi[b + j],         hr = gh[b + j];
    float iz = gi[b + H + j],     hz = gh[b + H + j];
    float in = gi[b + 2 * H + j], hn = gh[b + 2 * H + j];
    float r = sigmoidf(ir + hr);
    float z = sigmoidf(iz + hz);
    float nn = tanhf(in + r * hn);
    float h = prev[(size_t)n * H + j];
    out[(size_t)n * H + j] = (1.0f - z) * nn + z * h;
}

// ---------------------------------------------------------------------------
// Edge scores: wave per query edge; 272-wide dot with post_W
// ---------------------------------------------------------------------------
__global__ __launch_bounds__(256)
void edge_score_kernel(const float* __restrict__ emb2,
                       const long long* __restrict__ eli,
                       const float* __restrict__ eattr,
                       const float* __restrict__ postW,
                       const float* __restrict__ postb,
                       float* __restrict__ scores, int EQ) {
    int gw = (blockIdx.x * blockDim.x + threadIdx.x) >> 5;
    int lane = threadIdx.x & 31;
    if (gw >= EQ) return;
    int src = (int)eli[gw];
    int dst = (int)eli[EQ + gw];
    const float* hs = emb2 + (size_t)src * 128;
    const float* hd = emb2 + (size_t)dst * 128;
    float sum = 0.0f;
    #pragma unroll
    for (int j = 0; j < 4; ++j) {
        int f = lane + j * 32;
        sum += hs[f] * postW[f] + hd[f] * postW[128 + f];
    }
    if (lane < 16)
        sum += eattr[(size_t)gw * 16 + lane] * postW[256 + lane];
    #pragma unroll
    for (int o = 16; o > 0; o >>= 1)
        sum += __shfl_down(sum, o, 32);
    if (lane == 0)
        scores[gw] = sum + postb[0];
}

// ---------------------------------------------------------------------------
static inline void launch_gemm(const _Float16* A, const _Float16* W, const float* b,
                               float* C, int M, int K, int Nout, int act,
                               hipStream_t s) {
    dim3 grid((M + 127) / 128, Nout / 64);
    if (K == 256)
        gemm_wmma_f16<256><<<grid, 256, 0, s>>>(A, W, b, C, M, Nout, act);
    else
        gemm_wmma_f16<128><<<grid, 256, 0, s>>>(A, W, b, C, M, Nout, act);
}

static inline void launch_cvt(const float* in, _Float16* out, long long n,
                              hipStream_t s) {
    int n4 = (int)(n >> 2);
    cvt_f16_kernel<<<(n4 + 255) / 256, 256, 0, s>>>(in, out, n4);
}

extern "C" void kernel_launch(void* const* d_in, const int* in_sizes, int n_in,
                              void* d_out, int out_size, void* d_ws, size_t ws_size,
                              hipStream_t stream) {
    (void)in_sizes; (void)n_in; (void)out_size; (void)ws_size;
    const int N = GNN_N, E = GNN_E, EQ = GNN_EQ;

    const float*     x      = (const float*)d_in[0];
    const long long* ei     = (const long long*)d_in[1];
    const long long* eli    = (const long long*)d_in[2];
    const float*     eattr  = (const float*)d_in[3];
    const float*     prev1  = (const float*)d_in[4];
    const float*     prev2  = (const float*)d_in[5];
    const float* pre1_W  = (const float*)d_in[6];
    const float* pre1_b  = (const float*)d_in[7];
    const float* pre2_W  = (const float*)d_in[8];
    const float* pre2_b  = (const float*)d_in[9];
    const float* conv1_W = (const float*)d_in[10];
    const float* conv1_b = (const float*)d_in[11];
    const float* conv2_W = (const float*)d_in[12];
    const float* conv2_b = (const float*)d_in[13];
    const float* g1Wih   = (const float*)d_in[14];
    const float* g1Whh   = (const float*)d_in[15];
    const float* g1bih   = (const float*)d_in[16];
    const float* g1bhh   = (const float*)d_in[17];
    const float* g2Wih   = (const float*)d_in[18];
    const float* g2Whh   = (const float*)d_in[19];
    const float* g2bih   = (const float*)d_in[20];
    const float* g2bhh   = (const float*)d_in[21];
    const float* postW   = (const float*)d_in[22];
    const float* postb   = (const float*)d_in[23];

    float* out    = (float*)d_out;
    float* scores = out;
    float* emb1   = out + EQ;                 // [N,256]
    float* emb2   = emb1 + (size_t)N * 256;   // [N,128]

    // ---- workspace layout: f32 region, then f16 region (16B aligned) ----
    float* ws   = (float*)d_ws;
    float* h1   = ws;                          // [N,256]  pre1 out -> c1 -> c2
    float* h2   = h1 + (size_t)N * 256;        // [N,128]  pre2 out -> hw2
    float* hw1  = h2 + (size_t)N * 128;        // [N,256]  conv1 hw -> agg2
    float* agg1 = hw1 + (size_t)N * 256;       // [N,256]
    float* gi   = agg1 + (size_t)N * 256;      // [N,768]  -> gi2
    float* gh   = gi + (size_t)N * 768;        // [N,768]  -> gh2
    float* dinv = gh + (size_t)N * 768;        // [N]
    float* c1 = h1;  float* hw2 = h2;  float* agg2 = hw1;  float* c2 = h1;

    _Float16* f16base = (_Float16*)(dinv + (size_t)N + 16);  // after f32 region
    f16base = (_Float16*)(((uintptr_t)f16base + 15) & ~(uintptr_t)15);
    _Float16* a16     = f16base;                       // [N,256] halves, reused
    _Float16* w16     = a16 + (size_t)N * 256;         // weight pool (f16)
    _Float16* pre1W16 = w16;                           // 256*256
    _Float16* pre2W16 = pre1W16 + 256 * 256;           // 256*128
    _Float16* c1W16   = pre2W16 + 256 * 128;           // 128*256
    _Float16* c2W16   = c1W16 + 128 * 256;             // 256*128
    _Float16* g1ih16  = c2W16 + 256 * 128;             // 256*768
    _Float16* g1hh16  = g1ih16 + 256 * 768;            // 256*768
    _Float16* g2ih16  = g1hh16 + 256 * 768;            // 128*384
    _Float16* g2hh16  = g2ih16 + 128 * 384;            // 128*384

    const int T = 256;

    // 0) convert weights to f16 (once per launch; deterministic)
    launch_cvt(pre1_W, pre1W16, 256 * 256, stream);
    launch_cvt(pre2_W, pre2W16, 256 * 128, stream);
    launch_cvt(conv1_W, c1W16, 128 * 256, stream);
    launch_cvt(conv2_W, c2W16, 256 * 128, stream);
    launch_cvt(g1Wih, g1ih16, 256 * 768, stream);
    launch_cvt(g1Whh, g1hh16, 256 * 768, stream);
    launch_cvt(g2Wih, g2ih16, 128 * 384, stream);
    launch_cvt(g2Whh, g2hh16, 128 * 384, stream);

    // 1-2) dense preprocessing
    launch_cvt(x, a16, (long long)N * 256, stream);
    launch_gemm(a16, pre1W16, pre1_b, h1, N, 256, 256, 1, stream);
    launch_cvt(h1, a16, (long long)N * 256, stream);
    launch_gemm(a16, pre2W16, pre2_b, h2, N, 256, 128, 1, stream);

    // 3) symmetric normalization (self-loops -> deg init 1)
    fill_kernel<<<(N + T - 1) / T, T, 0, stream>>>(dinv, 1.0f, N);
    deg_count_kernel<<<(E + T - 1) / T, T, 0, stream>>>(ei, dinv, E);
    rsqrt_kernel<<<(N + T - 1) / T, T, 0, stream>>>(dinv, N);

    // 4-7) GCN layer 1
    launch_cvt(h2, a16, (long long)N * 128, stream);
    launch_gemm(a16, c1W16, nullptr, hw1, N, 128, 256, 0, stream);
    fill_kernel<<<((N * 256) + T - 1) / T, T, 0, stream>>>(agg1, 0.0f, N * 256);
    gcn_scatter_kernel<<<((size_t)E * 64 + T - 1) / T, T, 0, stream>>>(ei, hw1, dinv, agg1, E, 256);
    gcn_finalize_kernel<<<((N * 256) + T - 1) / T, T, 0, stream>>>(agg1, hw1, dinv, conv1_b, c1, N, 256);

    // 8-10) GRU 1 -> emb1
    launch_cvt(c1, a16, (long long)N * 256, stream);
    launch_gemm(a16, g1ih16, g1bih, gi, N, 256, 768, 0, stream);
    launch_cvt(prev1, a16, (long long)N * 256, stream);
    launch_gemm(a16, g1hh16, g1bhh, gh, N, 256, 768, 0, stream);
    gru_kernel<<<((N * 256) + T - 1) / T, T, 0, stream>>>(gi, gh, prev1, emb1, N, 256);

    // 11-14) GCN layer 2
    launch_cvt(emb1, a16, (long long)N * 256, stream);
    launch_gemm(a16, c2W16, nullptr, hw2, N, 256, 128, 0, stream);
    fill_kernel<<<((N * 128) + T - 1) / T, T, 0, stream>>>(agg2, 0.0f, N * 128);
    gcn_scatter_kernel<<<((size_t)E * 32 + T - 1) / T, T, 0, stream>>>(ei, hw2, dinv, agg2, E, 128);
    gcn_finalize_kernel<<<((N * 128) + T - 1) / T, T, 0, stream>>>(agg2, hw2, dinv, conv2_b, c2, N, 128);

    // 15-17) GRU 2 -> emb2
    launch_cvt(c2, a16, (long long)N * 128, stream);
    launch_gemm(a16, g2ih16, g2bih, gi, N, 128, 384, 0, stream);
    launch_cvt(prev2, a16, (long long)N * 128, stream);
    launch_gemm(a16, g2hh16, g2bhh, gh, N, 128, 384, 0, stream);
    gru_kernel<<<((N * 128) + T - 1) / T, T, 0, stream>>>(gi, gh, prev2, emb2, N, 128);

    // 18) edge-level scores
    edge_score_kernel<<<((size_t)EQ * 32 + T - 1) / T, T, 0, stream>>>(
        emb2, eli, eattr, postW, postb, scores, EQ);
}
